// SharedClassName_65481071405965
// MI455X (gfx1250) — matrix operand order repaired and checked by reading, not compiled, not run on hardware
//
#include <hip/hip_runtime.h>
#include <hip/hip_bf16.h>

// ---------------------------------------------------------------------------
// RBF-kernel features + 2-layer MLP, bf16 WMMA pipeline for gfx1250 (MI455X).
// ---------------------------------------------------------------------------

typedef __bf16 bf16_t;
typedef bf16_t v16bf __attribute__((ext_vector_type(16)));
typedef float  v8f   __attribute__((ext_vector_type(8)));

#define GAMMA_F 0.00390625f

union Frag32B { v16bf v; uint4 u[2]; };

__device__ __forceinline__ unsigned short f2bf(float f) {
  union { float f; unsigned int u; } c; c.f = f;
  unsigned int u = c.u;
  if ((u & 0x7fffffffu) > 0x7f800000u)          // NaN: quiet it
    return (unsigned short)((u >> 16) | 0x0040u);
  return (unsigned short)((u + 0x7fffu + ((u >> 16) & 1u)) >> 16);  // RNE
}

__device__ __forceinline__ float bf2f(unsigned short h) {
  union { float f; unsigned int u; } c; c.u = ((unsigned int)h) << 16;
  return c.f;
}

// --- f32 -> bf16 row convert + row sum-of-squares (one wave per row) -------
__global__ void prep_rows_kernel(const float* __restrict__ src,
                                 unsigned short* __restrict__ dst,
                                 float* __restrict__ sums,
                                 int rows, int cols) {
  const int row  = blockIdx.x * (blockDim.x >> 5) + (threadIdx.x >> 5);
  const int lane = threadIdx.x & 31;
  if (row >= rows) return;
  float s = 0.0f;
  const size_t base = (size_t)row * cols;
  for (int i = lane; i < cols; i += 32) {
    float v = src[base + i];
    s += v * v;
    dst[base + i] = f2bf(v);
  }
  for (int off = 16; off > 0; off >>= 1) s += __shfl_down(s, off, 32);
  if (lane == 0) sums[row] = s;
}

// --- plain f32 -> bf16 convert (4-wide) ------------------------------------
__global__ void convert_bf16_kernel(const float* __restrict__ src,
                                    unsigned short* __restrict__ dst,
                                    size_t n) {
  size_t i = ((size_t)blockIdx.x * blockDim.x + threadIdx.x) * 4;
  if (i + 3 < n) {
    float4 v = *(const float4*)(src + i);
    dst[i + 0] = f2bf(v.x);
    dst[i + 1] = f2bf(v.y);
    dst[i + 2] = f2bf(v.z);
    dst[i + 3] = f2bf(v.w);
  } else {
    for (; i < n; ++i) dst[i] = f2bf(src[i]);
  }
}

// --- WMMA GEMM: Out = epilogue(A (MxK) * W^T (KxN)) ------------------------
// A: M x K bf16 row-major.  W: N x K bf16 row-major (so B[k][n] = W[n][k]).
// EPI == 0: RBF  -> out = exp(-gamma * (rowterm[m] + colterm[n] - 2*dot))
// EPI == 1: ReLU -> out = max(dot + colterm[n], 0)
// Block = 256 threads (8 waves), tile 128x128, wave tile 32x64 (2x4 subtiles).
template <int EPI>
__global__ __launch_bounds__(256)
void gemm_wmma_kernel(const unsigned short* __restrict__ A,
                      const unsigned short* __restrict__ W,
                      unsigned short* __restrict__ Out,
                      const float* __restrict__ rowterm,
                      const float* __restrict__ colterm,
                      int M, int N, int K) {
  const int lane = threadIdx.x & 31;
  const int wid  = threadIdx.x >> 5;       // 0..7
  const int wm   = wid >> 1;               // 0..3 -> 32-row slabs
  const int wn   = wid & 1;                // 0..1 -> 64-col slabs

  const int m_base = blockIdx.y * 128 + wm * 32;
  const int n_base = blockIdx.x * 128 + wn * 64;

  const int hb  = lane >> 4;               // half-wave select
  const int l15 = lane & 15;

  v8f acc[2][4];
#pragma unroll
  for (int i = 0; i < 2; ++i)
#pragma unroll
    for (int j = 0; j < 4; ++j) acc[i][j] = v8f{};

  for (int kk = 0; kk < K; kk += 32) {
    Frag32B a[2], b[4];
    // A fragment: lane -> row m_sub + l15; elems 0..7 = K {kk+8*hb .. +7},
    // elems 8..15 = K {kk+16+8*hb .. +7}  (ISA 16-bit A 16x32 layout)
#pragma unroll
    for (int mi = 0; mi < 2; ++mi) {
      const unsigned short* p =
          A + (size_t)(m_base + mi * 16 + l15) * K + kk + hb * 8;
      a[mi].u[0] = *(const uint4*)(p);
      a[mi].u[1] = *(const uint4*)(p + 16);
    }
    // B fragment: lane -> col n_sub + l15; elems 0..7 = K {kk+8*hb .. +7},
    // elems 8..15 = K {kk+16+8*hb .. +7} (same order as A; W is (N, K) row-major).
#pragma unroll
    for (int ni = 0; ni < 4; ++ni) {
      const unsigned short* p =
          W + (size_t)(n_base + ni * 16 + l15) * K + kk + hb * 8;
      b[ni].u[0] = *(const uint4*)(p);
      b[ni].u[1] = *(const uint4*)(p + 16);
    }
#pragma unroll
    for (int mi = 0; mi < 2; ++mi)
#pragma unroll
      for (int ni = 0; ni < 4; ++ni)
        acc[mi][ni] = __builtin_amdgcn_wmma_f32_16x16x32_bf16(
            false, a[mi].v, false, b[ni].v, (short)0, acc[mi][ni],
            false, false);
  }

  // Epilogue + store (C/D layout: VGPR v, lane<16 -> M=v, lane>=16 -> M=v+8)
#pragma unroll
  for (int mi = 0; mi < 2; ++mi) {
#pragma unroll
    for (int ni = 0; ni < 4; ++ni) {
      const int col = n_base + ni * 16 + l15;
      const float ct = colterm[col];
#pragma unroll
      for (int v = 0; v < 8; ++v) {
        const int row = m_base + mi * 16 + hb * 8 + v;
        float r = acc[mi][ni][v];
        if (EPI == 0) {
          const float d2 = rowterm[row] + ct - 2.0f * r;
          r = __expf(-GAMMA_F * d2);
        } else {
          r = r + ct;
          r = r > 0.0f ? r : 0.0f;
        }
        Out[(size_t)row * N + col] = f2bf(r);
      }
    }
  }
}

// --- 2-logit head: one wave per row ----------------------------------------
__global__ void head_kernel(const unsigned short* __restrict__ h2,
                            const float* __restrict__ Wh,
                            const float* __restrict__ bh,
                            float* __restrict__ out, int S) {
  const int row  = blockIdx.x * (blockDim.x >> 5) + (threadIdx.x >> 5);
  const int lane = threadIdx.x & 31;
  float s0 = 0.0f, s1 = 0.0f;
  const size_t base = (size_t)row * S;
  for (int i = lane; i < S; i += 32) {
    const float h = bf2f(h2[base + i]);
    s0 += h * Wh[i];
    s1 += h * Wh[S + i];
  }
  for (int off = 16; off > 0; off >>= 1) {
    s0 += __shfl_down(s0, off, 32);
    s1 += __shfl_down(s1, off, 32);
  }
  if (lane == 0) {
    out[(size_t)row * 2 + 0] = s0 + bh[0];
    out[(size_t)row * 2 + 1] = s1 + bh[1];
  }
}

// ---------------------------------------------------------------------------
extern "C" void kernel_launch(void* const* d_in, const int* in_sizes, int n_in,
                              void* d_out, int out_size, void* d_ws,
                              size_t ws_size, hipStream_t stream) {
  const float* x  = (const float*)d_in[0];
  const float* sv = (const float*)d_in[1];
  const float* W1 = (const float*)d_in[2];
  const float* b1 = (const float*)d_in[3];
  const float* W2 = (const float*)d_in[4];
  const float* b2 = (const float*)d_in[5];
  const float* Wh = (const float*)d_in[6];
  const float* bh = (const float*)d_in[7];
  float* out = (float*)d_out;

  const int S = in_sizes[3];            // 1024 (b1 length)
  const int F = in_sizes[1] / S;        // 256  (sv is S x F)
  const int B = in_sizes[0] / F;        // 65536

  // workspace carve-out (256B aligned slices)
  char* w = (char*)d_ws;
  auto carve = [&](size_t bytes) -> void* {
    void* p = (void*)w;
    w += (bytes + 255) & ~(size_t)255;
    return p;
  };
  unsigned short* x_bf  = (unsigned short*)carve((size_t)B * F * 2);
  unsigned short* sv_bf = (unsigned short*)carve((size_t)S * F * 2);
  unsigned short* W1_bf = (unsigned short*)carve((size_t)S * S * 2);
  unsigned short* W2_bf = (unsigned short*)carve((size_t)S * S * 2);
  float*          x2    = (float*)carve((size_t)B * 4);
  float*          sv2   = (float*)carve((size_t)S * 4);
  unsigned short* bufA  = (unsigned short*)carve((size_t)B * S * 2);  // k, then h2
  unsigned short* bufB  = (unsigned short*)carve((size_t)B * S * 2);  // h1

  // 1) convert x, sv to bf16 + squared row norms
  prep_rows_kernel<<<B / 8, 256, 0, stream>>>(x, x_bf, x2, B, F);
  prep_rows_kernel<<<(S + 7) / 8, 256, 0, stream>>>(sv, sv_bf, sv2, S, F);

  // 2) convert weights
  const size_t nw = (size_t)S * S;
  convert_bf16_kernel<<<(unsigned)((nw / 4 + 255) / 256), 256, 0, stream>>>(W1, W1_bf, nw);
  convert_bf16_kernel<<<(unsigned)((nw / 4 + 255) / 256), 256, 0, stream>>>(W2, W2_bf, nw);

  // 3) k = exp(-gamma * (x2 + sv2 - 2 * x.sv^T))   (B x S, bf16)
  dim3 grid1(S / 128, B / 128);
  gemm_wmma_kernel<0><<<grid1, 256, 0, stream>>>(x_bf, sv_bf, bufA, x2, sv2, B, S, F);

  // 4) h1 = relu(k @ W1^T + b1)
  gemm_wmma_kernel<1><<<grid1, 256, 0, stream>>>(bufA, W1_bf, bufB, nullptr, b1, B, S, S);

  // 5) h2 = relu(h1 @ W2^T + b2)
  gemm_wmma_kernel<1><<<grid1, 256, 0, stream>>>(bufB, W2_bf, bufA, nullptr, b2, B, S, S);

  // 6) logits = h2 @ Wh^T + bh  (B x 2, f32)
  head_kernel<<<B / 8, 256, 0, stream>>>(bufA, Wh, bh, out, S);
}
